// QuantizedExpert_42176578847201
// MI455X (gfx1250) — matrix operand order repaired and checked by reading, not compiled, not run on hardware
//
#include <hip/hip_runtime.h>

// ---------------------------------------------------------------------------
// Trellis-quantized MLP expert for MI455X (gfx1250, wave32, WMMA, async-LDS).
// 1) dequant walks -> f16 weights (201MB read ~9us @23.3TB/s, one-time)
// 2) x f32 -> f16 once (so all GEMM staging is 16B f16 chunks)
// 3) fused gate/up GEMM + silu*mul (f16 WMMA, f32 accum), double-buffered LDS
//    fed by GLOBAL_LOAD_ASYNC_TO_LDS_B128 (ASYNCcnt pipeline)
// 4) down GEMM -> f32 out, same async double-buffered skeleton
// ~412 GFLOP total -> compute-bound; f16 weights (100MB) live in 192MB L2.
// ---------------------------------------------------------------------------

typedef __attribute__((ext_vector_type(16))) _Float16 v16h;
typedef __attribute__((ext_vector_type(8)))  float    v8f;
typedef __attribute__((ext_vector_type(4)))  _Float16 v4h;

#define D_MODEL 2048
#define D_FF    8192
#define N_TOK   4096
#define W_SCALE 0.02f

#define TM  128     // block tile M (tokens)
#define TN  128     // block tile N (output features)
#define TK  32      // K step (matches 16x16x32 f16 WMMA)
#define LDT 40      // padded LDS row stride in halves (80B, 16B-aligned rows)

// ---------------------------------------------------------------------------
// Async global->LDS 16B copy (gfx1250): 16 bytes at gaddr go to LDS at lds_addr.
// LDS byte address = low 32 bits of the generic pointer (== AS3 offset).
// Tracked by ASYNCcnt; completes without touching VGPRs.
// ---------------------------------------------------------------------------
__device__ __forceinline__ void async_ld_b128(unsigned lds_addr, const void* gaddr)
{
    asm volatile("global_load_async_to_lds_b128 %0, %1, off"
                 :: "v"(lds_addr), "v"(gaddr)
                 : "memory");
}

__device__ __forceinline__ void wait_async0()
{
#if __has_builtin(__builtin_amdgcn_s_wait_asynccnt)
    __builtin_amdgcn_s_wait_asynccnt(0);
#else
    asm volatile("s_wait_asynccnt 0" ::: "memory");
#endif
}

// ---------------------------------------------------------------------------
// Dequant: out[r,c] = (f16)(lut[walks[r,c]] * sign_l[r] * sign_r[c] * 0.02)
// ---------------------------------------------------------------------------
__global__ __launch_bounds__(256) void qe_dequant(
    const int* __restrict__ walks, const float* __restrict__ lut,
    const float* __restrict__ sl, const float* __restrict__ sr,
    _Float16* __restrict__ out, int cols)
{
    size_t idx4 = ((size_t)blockIdx.x * blockDim.x + threadIdx.x) * 4;
    int row = (int)(idx4 / (size_t)cols);
    int col = (int)(idx4 % (size_t)cols);
    const int4 w = *(const int4*)(walks + idx4);
    const float s = sl[row] * W_SCALE;
    v4h o;
    o[0] = (_Float16)(lut[w.x] * s * sr[col + 0]);
    o[1] = (_Float16)(lut[w.y] * s * sr[col + 1]);
    o[2] = (_Float16)(lut[w.z] * s * sr[col + 2]);
    o[3] = (_Float16)(lut[w.w] * s * sr[col + 3]);
    *(v4h*)(out + idx4) = o;
}

// ---------------------------------------------------------------------------
// x: f32 -> f16 (one cheap pass so GEMM staging is pure 16B async copies)
// ---------------------------------------------------------------------------
__global__ __launch_bounds__(256) void qe_cvt_x(
    const float* __restrict__ x, _Float16* __restrict__ xh)
{
    size_t i4 = ((size_t)blockIdx.x * blockDim.x + threadIdx.x) * 4;
    float4 v = *(const float4*)(x + i4);
    v4h o;
    o[0] = (_Float16)v.x; o[1] = (_Float16)v.y;
    o[2] = (_Float16)v.z; o[3] = (_Float16)v.w;
    *(v4h*)(xh + i4) = o;
}

// ---------------------------------------------------------------------------
// LDS fragment loaders per CDNA5 wave32 WMMA layouts (CDNA5 ISA).
// A 16x32 f16: M = lane&15; lanes 0-15 K {0..7,16..23}, lanes 16-31 {8..15,24..31}.
// B 32x16 f16: N = lane&15; lanes 0-15 K {0..7,16..23}, lanes 16-31 {8..15,24..31}.
// ---------------------------------------------------------------------------
__device__ __forceinline__ v16h lds_frag_a(const _Float16* base)
{
    const int lane = threadIdx.x & 31;
    const _Float16* p = base + (lane & 15) * LDT + ((lane >> 4) << 3);
    v16h a;
    ((uint4*)&a)[0] = *(const uint4*)(p);
    ((uint4*)&a)[1] = *(const uint4*)(p + 16);
    return a;
}

__device__ __forceinline__ v16h lds_frag_b(const _Float16* base)
{
    const int lane = threadIdx.x & 31;
    const _Float16* p = base + (lane & 15) * LDT + ((lane >> 4) << 3);
    v16h b;
    ((uint4*)&b)[0] = *(const uint4*)(p);
    ((uint4*)&b)[1] = *(const uint4*)(p + 16);
    return b;
}

// ---------------------------------------------------------------------------
// Fused gate+up GEMM: hidden = silu(xh Wg^T) * (xh Wu^T), f16 output.
// 8 waves (2x4); each wave owns 64x32 = 4x2 fragments for BOTH gate and up
// (16 WMMAs / wave / K-step). Double-buffered LDS, async-LDS staging.
// ---------------------------------------------------------------------------
__global__ __launch_bounds__(256) void qe_gateup(
    const _Float16* __restrict__ xh, const _Float16* __restrict__ Wg,
    const _Float16* __restrict__ Wu, _Float16* __restrict__ hidden)
{
    __shared__ _Float16 As[2][TM * LDT];
    __shared__ _Float16 Bg[2][TN * LDT];
    __shared__ _Float16 Bu[2][TN * LDT];

    const int tid   = threadIdx.x;
    const int lane  = tid & 31;
    const int wave  = tid >> 5;
    const int waveM = wave & 1;   // 0..1  -> 64-row slab
    const int waveN = wave >> 1;  // 0..3  -> 32-col slab

    const int mBase = blockIdx.y * TM;
    const int nBase = blockIdx.x * TN;

    // per-thread staging chunk: 2 x (row, 8-half col chunk) covers 128x32 tile
    const int r0  = tid >> 2;                 // rows 0..63
    const int c0  = (tid & 3) << 3;           // half-offset 0/8/16/24
    const unsigned loffA0 = (unsigned)(uintptr_t)&As[0][0];
    const unsigned loffG0 = (unsigned)(uintptr_t)&Bg[0][0];
    const unsigned loffU0 = (unsigned)(uintptr_t)&Bu[0][0];

    auto stage = [&](int buf, int k0) {
        #pragma unroll
        for (int i = 0; i < 2; ++i) {
            const int row = r0 + i * 64;
            const unsigned lo = (unsigned)(buf * TM * LDT + row * LDT + c0) * 2u;
            async_ld_b128(loffA0 + lo, xh + (size_t)(mBase + row) * D_MODEL + k0 + c0);
            async_ld_b128(loffG0 + lo, Wg + (size_t)(nBase + row) * D_MODEL + k0 + c0);
            async_ld_b128(loffU0 + lo, Wu + (size_t)(nBase + row) * D_MODEL + k0 + c0);
        }
    };

    v8f zero = {};
    v8f accG[4][2], accU[4][2];
    for (int fm = 0; fm < 4; ++fm)
        for (int fn = 0; fn < 2; ++fn) { accG[fm][fn] = zero; accU[fm][fn] = zero; }

    stage(0, 0);
    wait_async0();
    __syncthreads();

    const int nk = D_MODEL / TK;
    for (int it = 0; it < nk; ++it) {
        const int cur = it & 1;
        if (it + 1 < nk) stage(cur ^ 1, (it + 1) * TK);   // overlap with WMMA

        v16h a[4];
        #pragma unroll
        for (int fm = 0; fm < 4; ++fm)
            a[fm] = lds_frag_a(&As[cur][(waveM * 64 + fm * 16) * LDT]);
        #pragma unroll
        for (int fn = 0; fn < 2; ++fn) {
            v16h bg = lds_frag_b(&Bg[cur][(waveN * 32 + fn * 16) * LDT]);
            v16h bu = lds_frag_b(&Bu[cur][(waveN * 32 + fn * 16) * LDT]);
            #pragma unroll
            for (int fm = 0; fm < 4; ++fm) {
                accG[fm][fn] = __builtin_amdgcn_wmma_f32_16x16x32_f16(
                    false, a[fm], false, bg, (short)0, accG[fm][fn], false, false);
                accU[fm][fn] = __builtin_amdgcn_wmma_f32_16x16x32_f16(
                    false, a[fm], false, bu, (short)0, accU[fm][fn], false, false);
            }
        }
        wait_async0();       // next buffer's async writes landed in LDS
        __syncthreads();
    }

    // ---- epilogue: silu(gate) * up -> f16 hidden ----
    const int rOff = (lane >> 4) << 3;   // C/D layout: lanes 16-31 hold M+8
    const int nOff = lane & 15;
    #pragma unroll
    for (int fm = 0; fm < 4; ++fm)
        #pragma unroll
        for (int fn = 0; fn < 2; ++fn)
            #pragma unroll
            for (int r = 0; r < 8; ++r) {
                float g = accG[fm][fn][r];
                float u = accU[fm][fn][r];
                float h = (g / (1.0f + __expf(-g))) * u;
                int row = mBase + waveM * 64 + fm * 16 + rOff + r;
                int col = nBase + waveN * 32 + fn * 16 + nOff;
                hidden[(size_t)row * D_FF + col] = (_Float16)h;
            }
}

// ---------------------------------------------------------------------------
// Down projection: out = hidden Wd^T, f32 output. K = 8192.
// ---------------------------------------------------------------------------
__global__ __launch_bounds__(256) void qe_down(
    const _Float16* __restrict__ hidden, const _Float16* __restrict__ Wd,
    float* __restrict__ out)
{
    __shared__ _Float16 As[2][TM * LDT];
    __shared__ _Float16 Bs[2][TN * LDT];

    const int tid   = threadIdx.x;
    const int lane  = tid & 31;
    const int wave  = tid >> 5;
    const int waveM = wave & 1;
    const int waveN = wave >> 1;

    const int mBase = blockIdx.y * TM;
    const int nBase = blockIdx.x * TN;

    const int r0 = tid >> 2;
    const int c0 = (tid & 3) << 3;
    const unsigned loffA0 = (unsigned)(uintptr_t)&As[0][0];
    const unsigned loffB0 = (unsigned)(uintptr_t)&Bs[0][0];

    auto stage = [&](int buf, int k0) {
        #pragma unroll
        for (int i = 0; i < 2; ++i) {
            const int row = r0 + i * 64;
            const unsigned lo = (unsigned)(buf * TM * LDT + row * LDT + c0) * 2u;
            async_ld_b128(loffA0 + lo, hidden + (size_t)(mBase + row) * D_FF + k0 + c0);
            async_ld_b128(loffB0 + lo, Wd + (size_t)(nBase + row) * D_FF + k0 + c0);
        }
    };

    v8f zero = {};
    v8f acc[4][2];
    for (int fm = 0; fm < 4; ++fm)
        for (int fn = 0; fn < 2; ++fn) acc[fm][fn] = zero;

    stage(0, 0);
    wait_async0();
    __syncthreads();

    const int nk = D_FF / TK;
    for (int it = 0; it < nk; ++it) {
        const int cur = it & 1;
        if (it + 1 < nk) stage(cur ^ 1, (it + 1) * TK);

        v16h a[4];
        #pragma unroll
        for (int fm = 0; fm < 4; ++fm)
            a[fm] = lds_frag_a(&As[cur][(waveM * 64 + fm * 16) * LDT]);
        #pragma unroll
        for (int fn = 0; fn < 2; ++fn) {
            v16h b = lds_frag_b(&Bs[cur][(waveN * 32 + fn * 16) * LDT]);
            #pragma unroll
            for (int fm = 0; fm < 4; ++fm)
                acc[fm][fn] = __builtin_amdgcn_wmma_f32_16x16x32_f16(
                    false, a[fm], false, b, (short)0, acc[fm][fn], false, false);
        }
        wait_async0();
        __syncthreads();
    }

    const int rOff = (lane >> 4) << 3;
    const int nOff = lane & 15;
    #pragma unroll
    for (int fm = 0; fm < 4; ++fm)
        #pragma unroll
        for (int fn = 0; fn < 2; ++fn)
            #pragma unroll
            for (int r = 0; r < 8; ++r) {
                int row = mBase + waveM * 64 + fm * 16 + rOff + r;
                int col = nBase + waveN * 32 + fn * 16 + nOff;
                out[(size_t)row * D_MODEL + col] = acc[fm][fn][r];
            }
}

// ---------------------------------------------------------------------------
// Host launcher. ws layout (f16 elements):
//   [0, 16.8M)        Wg          [16.8M, 33.6M)  Wu
//   [33.6M, 50.3M)    Wd          [50.3M, 83.9M)  hidden [4096 x 8192]
//   [83.9M, 92.3M)    xh [4096 x 2048]            total ~185 MB
// ---------------------------------------------------------------------------
extern "C" void kernel_launch(void* const* d_in, const int* in_sizes, int n_in,
                              void* d_out, int out_size, void* d_ws, size_t ws_size,
                              hipStream_t stream)
{
    const float* x        = (const float*)d_in[0];
    const float* lut_g    = (const float*)d_in[1];
    const float* lut_u    = (const float*)d_in[2];
    const float* lut_d    = (const float*)d_in[3];
    const int*   walks_g  = (const int*)d_in[4];
    const int*   walks_u  = (const int*)d_in[5];
    const int*   walks_d  = (const int*)d_in[6];
    const float* sl_g     = (const float*)d_in[7];
    const float* sr_g     = (const float*)d_in[8];
    const float* sl_u     = (const float*)d_in[9];
    const float* sr_u     = (const float*)d_in[10];
    const float* sl_d     = (const float*)d_in[11];
    const float* sr_d     = (const float*)d_in[12];
    float* out            = (float*)d_out;

    const size_t SZ_W = (size_t)D_FF * D_MODEL;        // 16,777,216 elems
    const size_t SZ_H = (size_t)N_TOK * D_FF;          // 33,554,432 elems
    _Float16* ws     = (_Float16*)d_ws;
    _Float16* Wg     = ws;
    _Float16* Wu     = ws + SZ_W;
    _Float16* Wd     = ws + 2 * SZ_W;
    _Float16* hidden = ws + 3 * SZ_W;
    _Float16* xh     = ws + 3 * SZ_W + SZ_H;

    // 1) x -> f16
    qe_cvt_x<<<(int)((size_t)N_TOK * D_MODEL / 4 / 256), 256, 0, stream>>>(x, xh);

    // 2) dequantize all three weight matrices to f16
    const int dq_blocks = (int)(SZ_W / 4 / 256);       // 16384
    qe_dequant<<<dq_blocks, 256, 0, stream>>>(walks_g, lut_g, sl_g, sr_g, Wg, D_MODEL);
    qe_dequant<<<dq_blocks, 256, 0, stream>>>(walks_u, lut_u, sl_u, sr_u, Wu, D_MODEL);
    qe_dequant<<<dq_blocks, 256, 0, stream>>>(walks_d, lut_d, sl_d, sr_d, Wd, D_FF);

    // 3) fused gate/up GEMM + silu*mul -> hidden (f16)
    dim3 ggrid(D_FF / TN, N_TOK / TM);                 // (64, 32)
    qe_gateup<<<ggrid, 256, 0, stream>>>(xh, Wg, Wu, hidden);

    // 4) down projection -> out (f32)
    dim3 dgrid(D_MODEL / TN, N_TOK / TM);              // (16, 32)
    qe_down<<<dgrid, 256, 0, stream>>>(hidden, Wd, out);
}
